// BahdanauAttention_58617713656390
// MI455X (gfx1250) — matrix operand order repaired and checked by reading, not compiled, not run on hardware
//
#include <hip/hip_runtime.h>
#include <hip/hip_bf16.h>

// Problem constants (match reference)
static constexpr int kB = 32;
static constexpr int kS = 2048;
static constexpr int kH = 1024;
static constexpr int kU = 1024;

typedef __bf16 bf16;
typedef __attribute__((ext_vector_type(16))) __bf16 v16bf;
typedef __attribute__((ext_vector_type(4)))  __bf16 v4bf;
typedef __attribute__((ext_vector_type(8)))  float   v8f;

union Frag {
    v16bf v;
    uint4 q[2];
};

// ---------------------------------------------------------------------------
// Kernel 0: W1 [H,U] f32 row-major  ->  W1T [U,H] bf16 (K-contiguous per u)
// ---------------------------------------------------------------------------
__global__ void convert_w1_kernel(const float* __restrict__ W1, bf16* __restrict__ W1T) {
    const int g = blockIdx.x * 256 + threadIdx.x;   // g = u*kH + h (coalesced write)
    const int u = g >> 10;
    const int h = g & (kH - 1);
    W1T[g] = (bf16)W1[(long)h * kU + u];
}

// ---------------------------------------------------------------------------
// Kernel 1: proj_q[b,u] = query[b,:] @ W2[:,u] + b2[u]   (tiny: 67 MFLOP)
// ---------------------------------------------------------------------------
__global__ void projq_kernel(const float* __restrict__ query, const float* __restrict__ W2,
                             const float* __restrict__ b2, float* __restrict__ projq) {
    const int g = blockIdx.x * 256 + threadIdx.x;   // b*kU + u
    const int b = g >> 10;
    const int u = g & (kU - 1);
    float acc = b2[u];
    for (int h = 0; h < kH; ++h)
        acc += query[b * kH + h] * W2[(long)h * kU + u];   // coalesced over u
    projq[g] = acc;
}

// ---------------------------------------------------------------------------
// Kernel 2 (dominant): fused  score[b,s] = V . tanh(values[b,s,:]@W1 + b1 + projq[b,:])
// Grid: (kS/64, kB), 512 threads = 16 waves = 4 row-tiles x 4 u-groups.
// bf16 WMMA / f32 accumulate. A-chunk (HBM stream) is register-prefetched
// across the compute phase; B-slab (L2-resident W1T) is prefetched to WGP
// scope and copied after the barrier. LDS rows padded to 40 elems (80 B) ->
// conflict-free ds_load_b128; B-fragments double-buffered for partial waits.
// ---------------------------------------------------------------------------
__global__ __launch_bounds__(512) void score_kernel(
        const float* __restrict__ values, const bf16* __restrict__ W1T,
        const float* __restrict__ b1, const float* __restrict__ projq,
        const float* __restrict__ Vw, float* __restrict__ scores) {
    const int b    = blockIdx.y;
    const int s0   = blockIdx.x * 64;
    const int tid  = threadIdx.x;
    const int lane = tid & 31;
    const int wv   = tid >> 5;
    const int r    = wv & 3;          // row-tile within the 64-row block
    const int ug   = wv >> 2;         // u-group: 256 u-columns (16 tiles of 16)
    const int ubase = ug * 256;

    constexpr int LDB = 40;                        // padded LDS row stride (bf16 elems)
    __shared__ __align__(16) bf16 sB[kU * LDB];    // [u][k], 80 KB, W1 k-slab
    __shared__ __align__(16) bf16 sA[64 * LDB];    // [m][k], 5 KB, values tile
    __shared__ float sScore[64];

    v8f acc[16];
    #pragma unroll
    for (int ut = 0; ut < 16; ++ut)
        #pragma unroll
        for (int j = 0; j < 8; ++j) acc[ut][j] = 0.0f;

    const int am   = tid >> 3;                 // 0..63 : A row staged by this thread
    const int ap   = tid & 7;                  // 0..7  : float4 within 32-float chunk
    const int arow = r * 16 + (lane & 15);     // A-fragment row (M index)
    const int koff = (lane < 16) ? 0 : 8;      // A layout: low lanes K0..7/16..23, high K8..15/24..31
    const int bko  = (lane < 16) ? 0 : 8;      // B layout: low lanes K0..7/16..23, high K8..15/24..31

    const float* aSrc = values + ((long)(b * kS + s0 + am)) * kH + ap * 4;

    // ---- prologue: stage k0 = 0 directly ----
    {
        const float4 vs = *(const float4*)(aSrc);
        v4bf a4;
        a4.x = (bf16)vs.x; a4.y = (bf16)vs.y; a4.z = (bf16)vs.z; a4.w = (bf16)vs.w;
        *(v4bf*)(&sA[am * LDB + ap * 4]) = a4;
    }
    #pragma unroll
    for (int j = 0; j < 8; ++j) {
        const int i = tid + j * 512, u = i >> 2, p = i & 3;
        *(uint4*)(&sB[u * LDB + p * 8]) = *(const uint4*)(W1T + (long)u * kH + p * 8);
    }
    __syncthreads();

    float4 pA;   // register-staged next A chunk (HBM stream; only 4 VGPRs)

    for (int k0 = 0; k0 < kH; k0 += 32) {
        const bool more = (k0 + 32) < kH;
        if (more) {
            // A: issue HBM load now; consumed after the barrier below.
            pA = *(const float4*)(aSrc + k0 + 32);
            // B: W1T slab is L2-resident; pull next chunk into WGP-level cache
            // (locality 3 -> WGP scope, all cache levels). One lane per 64B row
            // chunk to avoid redundant line requests.
            #pragma unroll
            for (int j = 0; j < 8; ++j) {
                const int i = tid + j * 512, u = i >> 2, p = i & 3;
                if (p == 0)
                    __builtin_prefetch(W1T + (long)u * kH + (k0 + 32), 0, 3);
            }
        }

        // ---- compute on current LDS contents ----
        Frag fa;
        fa.q[0] = *(const uint4*)(&sA[arow * LDB + koff]);
        fa.q[1] = *(const uint4*)(&sA[arow * LDB + 16 + koff]);

        Frag fb0, fb1;
        {
            const int ub = (ubase + (lane & 15)) * LDB + bko;
            fb0.q[0] = *(const uint4*)(&sB[ub]);
            fb0.q[1] = *(const uint4*)(&sB[ub + 16]);
        }
        #pragma unroll
        for (int ut = 0; ut < 16; ++ut) {
            Frag& cur = (ut & 1) ? fb1 : fb0;
            if (ut < 15) {   // software-pipelined B-fragment prefetch
                Frag& nxt = (ut & 1) ? fb0 : fb1;
                const int ub = (ubase + (ut + 1) * 16 + (lane & 15)) * LDB + bko;
                nxt.q[0] = *(const uint4*)(&sB[ub]);
                nxt.q[1] = *(const uint4*)(&sB[ub + 16]);
            }
            acc[ut] = __builtin_amdgcn_wmma_f32_16x16x32_bf16(
                false, fa.v, false, cur.v, (short)0, acc[ut], false, false);
        }

        __syncthreads();   // all waves done reading sA/sB
        if (more) {
            // A chunk was loaded before the compute phase -> latency hidden.
            v4bf a4;
            a4.x = (bf16)pA.x; a4.y = (bf16)pA.y; a4.z = (bf16)pA.z; a4.w = (bf16)pA.w;
            *(v4bf*)(&sA[am * LDB + ap * 4]) = a4;
            // B slab copy: loads should hit WGP$/L2 after the prefetch above.
            #pragma unroll
            for (int j = 0; j < 8; ++j) {
                const int i = tid + j * 512, u = i >> 2, p = i & 3;
                *(uint4*)(&sB[u * LDB + p * 8]) =
                    *(const uint4*)(W1T + (long)u * kH + (k0 + 32) + p * 8);
            }
        }
        __syncthreads();
    }

    // --- Epilogue: + b1 + projq, tanh, dot with V, reduce to score per row ---
    if (tid < 64) sScore[tid] = 0.0f;
    __syncthreads();

    const int n  = lane & 15;         // N index within 16x16 tile
    const int hs = lane >> 4;         // which M-half this lane holds
    float sc[8];
    #pragma unroll
    for (int j = 0; j < 8; ++j) sc[j] = 0.0f;

    #pragma unroll
    for (int ut = 0; ut < 16; ++ut) {
        const int u = ubase + ut * 16 + n;
        const float bq = b1[u] + projq[b * kU + u];
        const float vv = Vw[u];
        #pragma unroll
        for (int j = 0; j < 8; ++j)
            sc[j] += tanhf(acc[ut][j] + bq) * vv;
    }
    // reduce over the 16 N-lanes (stays within each 16-lane half)
    #pragma unroll
    for (int j = 0; j < 8; ++j)
        for (int off = 1; off < 16; off <<= 1)
            sc[j] += __shfl_xor(sc[j], off, 32);
    if (n == 0) {
        #pragma unroll
        for (int j = 0; j < 8; ++j)
            atomicAdd(&sScore[r * 16 + j + 8 * hs], sc[j]);   // ds_add_f32 across u-groups
    }
    __syncthreads();
    if (tid < 64) scores[b * kS + s0 + tid] = sScore[tid];
}

// ---------------------------------------------------------------------------
// Kernel 3: softmax over S per batch (bv cancels in softmax)
// ---------------------------------------------------------------------------
__global__ void softmax_kernel(const float* __restrict__ scores, float* __restrict__ attn) {
    const int b = blockIdx.x;
    const int tid = threadIdx.x;
    __shared__ float sE[kS];
    __shared__ float sred[8];
    __shared__ float sbc;

    float m = -3.4e38f;
    for (int s = tid; s < kS; s += 256) {
        const float v = scores[b * kS + s];
        sE[s] = v;
        m = fmaxf(m, v);
    }
    for (int off = 16; off; off >>= 1) m = fmaxf(m, __shfl_xor(m, off, 32));
    if ((tid & 31) == 0) sred[tid >> 5] = m;
    __syncthreads();
    if (tid == 0) {
        float mm = sred[0];
        for (int i = 1; i < 8; ++i) mm = fmaxf(mm, sred[i]);
        sbc = mm;
    }
    __syncthreads();
    m = sbc;

    float sum = 0.0f;
    for (int s = tid; s < kS; s += 256) {
        const float e = __expf(sE[s] - m);
        sE[s] = e;
        sum += e;
    }
    for (int off = 16; off; off >>= 1) sum += __shfl_xor(sum, off, 32);
    __syncthreads();                 // protect sred reuse
    if ((tid & 31) == 0) sred[tid >> 5] = sum;
    __syncthreads();
    if (tid == 0) {
        float ss = 0.0f;
        for (int i = 0; i < 8; ++i) ss += sred[i];
        sbc = 1.0f / ss;
    }
    __syncthreads();
    const float inv = sbc;
    for (int s = tid; s < kS; s += 256) attn[b * kS + s] = sE[s] * inv;
}

// ---------------------------------------------------------------------------
// Kernel 4: context[b,h] = sum_s attn[b,s] * values[b,s,h]
// Grid (kB, kH/128), 128 threads; attn chunks broadcast via LDS.
// ---------------------------------------------------------------------------
__global__ void context_kernel(const float* __restrict__ values, const float* __restrict__ attn,
                               float* __restrict__ ctx) {
    const int b = blockIdx.x;
    const int h = blockIdx.y * 128 + threadIdx.x;
    __shared__ float w[128];
    float acc = 0.0f;
    for (int s0 = 0; s0 < kS; s0 += 128) {
        __syncthreads();
        w[threadIdx.x] = attn[b * kS + s0 + threadIdx.x];
        __syncthreads();
        #pragma unroll 4
        for (int i = 0; i < 128; ++i)
            acc += w[i] * values[((long)(b * kS) + s0 + i) * kH + h];   // coalesced over h
    }
    ctx[b * kH + h] = acc;
}

// ---------------------------------------------------------------------------
extern "C" void kernel_launch(void* const* d_in, const int* in_sizes, int n_in,
                              void* d_out, int out_size, void* d_ws, size_t ws_size,
                              hipStream_t stream) {
    const float* query  = (const float*)d_in[0];
    const float* values = (const float*)d_in[1];
    const float* W1     = (const float*)d_in[2];
    const float* b1     = (const float*)d_in[3];
    const float* W2     = (const float*)d_in[4];
    const float* b2     = (const float*)d_in[5];
    const float* Vw     = (const float*)d_in[6];
    // d_in[7] = bv: a uniform additive shift before softmax -> cancels; unused.
    (void)in_sizes; (void)n_in; (void)out_size; (void)ws_size;

    // Workspace layout: [W1T bf16: 2MB][projq f32: 128KB][scores f32: 256KB]
    bf16*  W1T    = (bf16*)d_ws;
    float* projq  = (float*)((char*)d_ws + (size_t)kU * kH * sizeof(bf16));
    float* scores = projq + (size_t)kB * kU;

    float* ctx  = (float*)d_out;            // [B,H] context first
    float* attn = ctx + (size_t)kB * kH;    // then [B,S,1] attention weights

    convert_w1_kernel<<<(kH * kU) / 256, 256, 0, stream>>>(W1, W1T);
    projq_kernel<<<(kB * kU) / 256, 256, 0, stream>>>(query, W2, b2, projq);
    score_kernel<<<dim3(kS / 64, kB), 512, 0, stream>>>(values, W1T, b1, projq, Vw, scores);
    softmax_kernel<<<kB, 256, 0, stream>>>(scores, attn);
    context_kernel<<<dim3(kB, kH / 128), 128, 0, stream>>>(values, attn, ctx);
}
